// FlexAttentionWrapper_4492535792468
// MI455X (gfx1250) — matrix operand order repaired and checked by reading, not compiled, not run on hardware
//
#include <hip/hip_runtime.h>

typedef _Float16 v16h __attribute__((ext_vector_type(16)));
typedef _Float16 v8h  __attribute__((ext_vector_type(8)));
typedef _Float16 v4h  __attribute__((ext_vector_type(4)));
typedef float    v8f  __attribute__((ext_vector_type(8)));

#define BB 2
#define HH 16
#define SS 2048
#define DD 128
#define BM 64          // query rows per block
#define TN 32          // kv tile size
#define PK 136         // K tile LDS pitch (halves): 128 + 8 pad
#define PV 40          // V^T / P tile LDS pitch (halves): 32 + 8 pad

#if defined(__gfx1250__) && __has_builtin(__builtin_amdgcn_global_load_async_to_lds_b128) && \
    __has_builtin(__builtin_amdgcn_s_wait_asynccnt)
#define USE_ASYNC 1
// exact parameter types per hipcc diagnostic: int4 in addrspace(1) / addrspace(3)
typedef int a4i __attribute__((vector_size(16)));
typedef a4i __attribute__((address_space(1)))* g4i_ptr;   // global (printed as __device__)
typedef a4i __attribute__((address_space(3)))* l4i_ptr;   // LDS (printed as __shared__)
#else
#define USE_ASYNC 0
#endif

static __device__ __forceinline__ v8f wmma16(v16h a, v16h b, v8f c) {
    // v_wmma_f32_16x16x32_f16: D = A(16x32 f16) * B(32x16 f16) + C(16x16 f32)
    return __builtin_amdgcn_wmma_f32_16x16x32_f16(false, a, false, b, (short)0, c, false, false);
}

static __device__ __forceinline__ v16h lds_v16(const _Float16* p) {
    v8h lo = *(const v8h*)p;        // 16B aligned -> ds_load_b128
    v8h hi = *(const v8h*)(p + 16);
    return __builtin_shufflevector(lo, hi, 0,1,2,3,4,5,6,7,8,9,10,11,12,13,14,15);
}

static __device__ __forceinline__ v4h cvt4(float x, float y, float z, float w) {
#if defined(__AMDGCN__) && __has_builtin(__builtin_amdgcn_cvt_pkrtz)
    auto a = __builtin_amdgcn_cvt_pkrtz(x, y);   // v_cvt_pk_rtz_f16_f32
    auto b = __builtin_amdgcn_cvt_pkrtz(z, w);
    v4h r;
    r[0] = (_Float16)a[0]; r[1] = (_Float16)a[1];
    r[2] = (_Float16)b[0]; r[3] = (_Float16)b[1];
    return r;
#else
    v4h r = { (_Float16)x, (_Float16)y, (_Float16)z, (_Float16)w };
    return r;
#endif
}

__global__ __launch_bounds__(128) void fattn_kernel(const float* __restrict__ Q,
                                                    const float* __restrict__ K,
                                                    const float* __restrict__ V,
                                                    float* __restrict__ O) {
    // raw f32 double buffers (async landing zone), f16 operand tiles
    __shared__ __align__(16) float    rawK[2][TN * DD];   // 2 x 16 KB
    __shared__ __align__(16) float    rawV[2][TN * DD];   // 2 x 16 KB
    __shared__ __align__(16) _Float16 sK[TN * PK];        // K tile, row-major [kv][d]
    __shared__ __align__(16) _Float16 sV[DD * PV];        // V tile, transposed [d][kv]
    __shared__ __align__(16) _Float16 sP[4 * 16 * PV];    // per-wave P tile [m][kv]

    const int tid  = threadIdx.x;
    const int wave = tid >> 5;
    const int lane = tid & 31;
    const int n15  = lane & 15;
    const int hi   = lane >> 4;

    const int qblk = blockIdx.x * BM;
    const int bh   = blockIdx.y;
    const size_t base = (size_t)bh * SS * DD;
    const float* Qb = Q + base;
    const float* Kb = K + base;
    const float* Vb = V + base;
    float*       Ob = O + base;

    const int q0w  = qblk + wave * 16;
    const int qrow = q0w + n15;

    // ---- issue one kv tile's K/V global->LDS copies (async if available) ----
    auto issue_tile = [&](int kvb, int buf) {
#pragma unroll
        for (int i = 0; i < 8; ++i) {
            const int row = wave * 8 + i;                 // one 512B row per instruction
            const float* gk = Kb + (size_t)(kvb + row) * DD + lane * 4;
            const float* gv = Vb + (size_t)(kvb + row) * DD + lane * 4;
            float* lk = &rawK[buf][row * DD + lane * 4];
            float* lv = &rawV[buf][row * DD + lane * 4];
#if USE_ASYNC
            __builtin_amdgcn_global_load_async_to_lds_b128((g4i_ptr)gk, (l4i_ptr)lk, 0, 0);
            __builtin_amdgcn_global_load_async_to_lds_b128((g4i_ptr)gv, (l4i_ptr)lv, 0, 0);
#else
            *(float4*)lk = *(const float4*)gk;
            *(float4*)lv = *(const float4*)gv;
#endif
        }
    };

    // ---- load Q tile into registers in WMMA A-layout, pre-scaled ----
    // A layout (16-bit, 16x32): lane = m + 16*((k>>3)&1); half = (k&7) + 8*(k>=16)
    const float qs = 0.088388347648318447f * 1.44269504088896340f; // 1/sqrt(D) * log2(e)
    v16h qa[4];
#pragma unroll
    for (int c = 0; c < 4; ++c) {
        const float* qp = Qb + (size_t)qrow * DD + 32 * c + hi * 8;
        float4 f0 = *(const float4*)(qp);
        float4 f1 = *(const float4*)(qp + 4);
        float4 f2 = *(const float4*)(qp + 16);
        float4 f3 = *(const float4*)(qp + 20);
        v4h h0 = cvt4(f0.x * qs, f0.y * qs, f0.z * qs, f0.w * qs);
        v4h h1 = cvt4(f1.x * qs, f1.y * qs, f1.z * qs, f1.w * qs);
        v4h h2 = cvt4(f2.x * qs, f2.y * qs, f2.z * qs, f2.w * qs);
        v4h h3 = cvt4(f3.x * qs, f3.y * qs, f3.z * qs, f3.w * qs);
#pragma unroll
        for (int j = 0; j < 4; ++j) {
            qa[c][j]      = h0[j];
            qa[c][4 + j]  = h1[j];
            qa[c][8 + j]  = h2[j];
            qa[c][12 + j] = h3[j];
        }
    }

    // ---- accumulators / softmax state ----
    v8f acc[8];
#pragma unroll
    for (int i = 0; i < 8; ++i)
#pragma unroll
        for (int j = 0; j < 8; ++j) acc[i][j] = 0.0f;
    float mi[8], li[8];
#pragma unroll
    for (int r = 0; r < 8; ++r) { mi[r] = -1.0e30f; li[r] = 0.0f; }

    _Float16* pw = &sP[wave * 16 * PV];
    const int ntiles = qblk / TN + 2;       // causal: kv up to qblk + BM - 1

    issue_tile(0, 0);                       // prefetch tile 0

    for (int t = 0; t < ntiles; ++t) {
        const int kvb  = t * TN;
        const bool more = (t + 1 < ntiles);
        if (more) issue_tile(kvb + TN, (t + 1) & 1);   // prefetch next tile
#if USE_ASYNC
        if (more) __builtin_amdgcn_s_wait_asynccnt(16);  // keep only next tile in flight
        else      __builtin_amdgcn_s_wait_asynccnt(0);
#endif
        __syncthreads();                    // all waves' copies for tile t landed

        // ---- convert pass: raw f32 LDS -> f16 K (row-major) + V^T ----
        const float* rk = rawK[t & 1];
        const float* rv = rawV[t & 1];
#pragma unroll
        for (int i = 0; i < 8; ++i) {
            int f   = tid + i * 128;
            int row = f >> 5;
            int c4  = f & 31;
            float4 kx = *(const float4*)(rk + row * DD + c4 * 4);
            *(v4h*)&sK[row * PK + c4 * 4] = cvt4(kx.x, kx.y, kx.z, kx.w);
            float4 vx = *(const float4*)(rv + row * DD + c4 * 4);
            v4h vh = cvt4(vx.x, vx.y, vx.z, vx.w);
            sV[(c4 * 4 + 0) * PV + row] = vh[0];
            sV[(c4 * 4 + 1) * PV + row] = vh[1];
            sV[(c4 * 4 + 2) * PV + row] = vh[2];
            sV[(c4 * 4 + 3) * PV + row] = vh[3];
        }
        __syncthreads();

        const bool active = (kvb <= q0w + 15);   // wave-uniform
        if (active) {
            // ---- scores: S = Q * K^T (two 16x16 tiles, K-dim = 128) ----
            // B layout (16-bit, 32x16): lane = n + 16*((k>>3)&1); half = (k&7) + 8*(k>=16)
            v8f s0, s1;
#pragma unroll
            for (int j = 0; j < 8; ++j) { s0[j] = 0.0f; s1[j] = 0.0f; }
#pragma unroll
            for (int c = 0; c < 4; ++c) {
                v16h b0 = lds_v16(&sK[n15 * PK + 32 * c + 8 * hi]);
                s0 = wmma16(qa[c], b0, s0);
            }
#pragma unroll
            for (int c = 0; c < 4; ++c) {
                v16h b1 = lds_v16(&sK[(16 + n15) * PK + 32 * c + 8 * hi]);
                s1 = wmma16(qa[c], b1, s1);
            }

            // ---- causal mask (diagonal tiles only; wave-uniform branch) ----
            if (kvb + TN - 1 > q0w) {
#pragma unroll
                for (int r = 0; r < 8; ++r) {
                    int qi = q0w + r + 8 * hi;          // C layout: M = r + 8*hi
                    if (kvb + n15 > qi)      s0[r] = -1.0e30f;
                    if (kvb + 16 + n15 > qi) s1[r] = -1.0e30f;
                }
            }

            // ---- online softmax (exp2 domain) ----
            float alph[8], p0[8], p1[8];
#pragma unroll
            for (int r = 0; r < 8; ++r) {
                float mx = fmaxf(s0[r], s1[r]);
                mx = fmaxf(mx, __shfl_xor(mx, 1, 32));
                mx = fmaxf(mx, __shfl_xor(mx, 2, 32));
                mx = fmaxf(mx, __shfl_xor(mx, 4, 32));
                mx = fmaxf(mx, __shfl_xor(mx, 8, 32));
                float mn = fmaxf(mi[r], mx);
                alph[r] = exp2f(mi[r] - mn);
                p0[r]   = exp2f(s0[r] - mn);
                p1[r]   = exp2f(s1[r] - mn);
                float rs = p0[r] + p1[r];
                rs += __shfl_xor(rs, 1, 32);
                rs += __shfl_xor(rs, 2, 32);
                rs += __shfl_xor(rs, 4, 32);
                rs += __shfl_xor(rs, 8, 32);
                li[r] = li[r] * alph[r] + rs;
                mi[r] = mn;
            }
#pragma unroll
            for (int dc = 0; dc < 8; ++dc)
#pragma unroll
                for (int r = 0; r < 8; ++r) acc[dc][r] *= alph[r];

            // ---- store P (C layout -> row-major LDS [m][kv]) ----
#pragma unroll
            for (int r = 0; r < 8; ++r) {
                int m = r + 8 * hi;
                pw[m * PV + n15]      = (_Float16)p0[r];
                pw[m * PV + 16 + n15] = (_Float16)p1[r];
            }
        }
        __syncthreads();                    // P write -> read; also V^T reuse fence

        if (active) {
            // ---- P in A layout: halves 0-7 = P[m][hi*8..], 8-15 = P[m][16+hi*8..]
            v8h plo = *(const v8h*)&pw[n15 * PV + hi * 8];
            v8h phi = *(const v8h*)&pw[n15 * PV + 16 + hi * 8];
            v16h pa = __builtin_shufflevector(plo, phi,
                        0,1,2,3,4,5,6,7,8,9,10,11,12,13,14,15);
            // ---- O += P * V   (8 chunks of 16 columns, K-dim = 32) ----
#pragma unroll
            for (int dc = 0; dc < 8; ++dc) {
                v16h bv = lds_v16(&sV[(16 * dc + n15) * PV + 8 * hi]);
                acc[dc] = wmma16(pa, bv, acc[dc]);
            }
        }
    }

    // ---- epilogue: O = acc / l ----
    float inv[8];
#pragma unroll
    for (int r = 0; r < 8; ++r) inv[r] = 1.0f / li[r];
#pragma unroll
    for (int dc = 0; dc < 8; ++dc) {
#pragma unroll
        for (int r = 0; r < 8; ++r) {
            int m = r + 8 * hi;
            Ob[(size_t)(q0w + m) * DD + dc * 16 + n15] = acc[dc][r] * inv[r];
        }
    }
}

extern "C" void kernel_launch(void* const* d_in, const int* in_sizes, int n_in,
                              void* d_out, int out_size, void* d_ws, size_t ws_size,
                              hipStream_t stream) {
    const float* q = (const float*)d_in[0];
    const float* k = (const float*)d_in[1];
    const float* v = (const float*)d_in[2];
    float* out = (float*)d_out;
    dim3 grid(SS / BM, BB * HH);
    dim3 block(128);
    hipLaunchKernelGGL(fattn_kernel, grid, block, 0, stream, q, k, v, out);
}
